// HierarchicalJointModelDAN_5128190951513
// MI455X (gfx1250) — matrix operand order repaired and checked by reading, not compiled, not run on hardware
//
#include <hip/hip_runtime.h>
#include <cstddef>

// ---------------- problem constants (from reference) ----------------
#define RR 64     // reviews
#define SS 32     // sentences / review
#define TT 50     // tokens / sentence
#define EE 512    // embed dim
#define UU 20     // user feats
#define LL 9      // labels
#define DD 532    // E + U

typedef __attribute__((ext_vector_type(16))) _Float16 v16h;
typedef __attribute__((ext_vector_type(8)))  _Float16 v8h;
typedef __attribute__((ext_vector_type(8)))  float    v8f;
typedef __attribute__((ext_vector_type(4)))  unsigned u32x4;
typedef __attribute__((ext_vector_type(8)))  unsigned u32x8;

__device__ __forceinline__ float selu_f(float x) {
    const float scale = 1.0507009873554805f;
    const float alpha = 1.6732632423543772f;
    return x > 0.f ? scale * x : scale * alpha * (__expf(x) - 1.f);
}

// Raw LDS byte offset of a generic pointer to a __shared__ object
// (generic -> addrspace(3) cast, then ptrtoint => LDS-relative address,
//  as required by async-load VDST and the TDM descriptor lds_addr).
__device__ __forceinline__ unsigned lds_off(void* p) {
    return (unsigned)(size_t)(__attribute__((address_space(3))) void*)p;
}

// ---------------------------------------------------------------
// Tensor Data Mover: stage one 16-row x 512-half f16 panel
// (16 KB, row stride 512 halves) from global into LDS.
// D# per CDNA5 ISA (2D tile, VADDR2/3 = NULL).
// Issued by one wave; completion via S_WAIT_TENSORCNT.
// ---------------------------------------------------------------
__device__ __forceinline__ void tdm_load_panel16(const _Float16* gsrc, void* ldsdst) {
    unsigned long long ga = (unsigned long long)(size_t)gsrc;
    u32x4 g0;
    g0[0] = 1u;                                  // count=1, user descriptor
    g0[1] = lds_off(ldsdst);                     // lds_addr (bytes)
    g0[2] = (unsigned)ga;                        // global_addr[31:0]
    g0[3] = (unsigned)((ga >> 32) & 0x01FFFFFFu) // global_addr[56:32]
          | (2u << 30);                          // type = 2 ("image")
    u32x8 g1;
    g1[0] = 0x00010000u;                         // workgroup_mask=0, data_size=1 (2B)
    g1[1] = 512u << 16;                          // tensor_dim0 = 512 (low 16 bits)
    g1[2] = 512u << 16;                          // tensor_dim0 hi=0 | tensor_dim1 lo = 512
    g1[3] = 512u << 16;                          // tensor_dim1 hi=0 | tile_dim0  = 512
    g1[4] = 16u;                                 // tile_dim1 = 16, tile_dim2 = 0
    g1[5] = 512u;                                // tensor_dim0_stride = 512 (low 32)
    g1[6] = 0u;                                  // stride hi / dim1_stride lo
    g1[7] = 0u;
    asm volatile("tensor_load_to_lds %0, %1" :: "s"(g0), "s"(g1) : "memory");
}

// ---------------------------------------------------------------
// Kernel 1: transpose + convert a 512x512 f32 weight to f16 [N][K].
// The 32x32 f32 source tile is staged with ASYNC global->LDS copies
// (one b128 chunk per thread), then converted/transposed out of LDS.
// Row pitch 36 floats keeps every 16B async destination aligned.
// ---------------------------------------------------------------
__global__ void k_convert_w(const float* __restrict__ w, _Float16* __restrict__ wt) {
    __shared__ float tf[32][36];
    int b   = blockIdx.x;          // 16 x 16 tiles
    int tk  = (b & 15) * 32;       // K tile base
    int tn  = (b >> 4) * 32;       // N tile base
    int t   = threadIdx.x;

    // async stage: thread t copies 16B of row (t>>3), segment (t&7)
    {
        int rr  = t >> 3;          // 0..31 source row (K)
        int seg = t & 7;           // 16B segment along N
        unsigned lofs = lds_off(&tf[rr][seg * 4]);
        unsigned long long ga =
            (unsigned long long)(size_t)(w + (size_t)(tk + rr) * EE + tn + seg * 4);
        asm volatile("global_load_async_to_lds_b128 %0, %1, off"
                     :: "v"(lofs), "v"(ga) : "memory");
        asm volatile("s_wait_asynccnt 0x0" ::: "memory");
    }
    __syncthreads();

    // convert + transpose out of LDS: wt[n][k] = (f16) w[k][n]
    int col = t & 31;              // k within tile
    int row = t >> 5;              // 0..7, n within tile (x4 iters)
#pragma unroll
    for (int i = 0; i < 4; ++i)
        wt[(size_t)(tn + row + i * 8) * EE + (tk + col)] =
            (_Float16)tf[col][row + i * 8];
}

// ---------------------------------------------------------------
// Kernel 2: embedding gather + masked mean over tokens.
// One block per (review,sentence); 256 lanes x float2 cover E=512.
// Token ids are block-uniform -> scalar loads; prefetch next row.
// ---------------------------------------------------------------
__global__ void k_tok_mean(const int* __restrict__ toks, const float* __restrict__ emb,
                           _Float16* __restrict__ smean, float* __restrict__ smask) {
    int rs = blockIdx.x;                 // 0..R*S-1
    int c  = threadIdx.x * 2;            // two consecutive columns
    const int* tp = toks + (size_t)rs * TT;
    float a0 = 0.f, a1 = 0.f;
    int cnt = 0;
    for (int t = 0; t < TT; ++t) {
        int tok = tp[t];
        if (t + 1 < TT)
            __builtin_prefetch(emb + (size_t)tp[t + 1] * EE + c, 0, 1);
        if (tok != 0) {
            const float2 e2 = *(const float2*)(emb + (size_t)tok * EE + c);
            a0 += e2.x; a1 += e2.y;
            ++cnt;
        }
    }
    float inv = 1.f / (float)(cnt > 0 ? cnt : 1);
    smean[(size_t)rs * EE + c]     = (_Float16)(a0 * inv);
    smean[(size_t)rs * EE + c + 1] = (_Float16)(a1 * inv);
    if (threadIdx.x == 0) smask[rs] = (cnt > 0) ? 1.f : 0.f;
}

// ---------------------------------------------------------------
// WMMA 16x16 tile: A fragment from global (row-major [M,K] f16),
// B fragment from the LDS panel (rows = output columns, [16][K]).
// Contiguous k order, same for A and B: lane l (g=l>>4, r=l&15):
//   A elements e     -> K = k0 + g*16 + e       (one 32B load)
//   B elements e     -> K = k0 + g*16 + e       (one 32B LDS load)
// ---------------------------------------------------------------
__device__ __forceinline__ v8f wmma_tile_mixed(const _Float16* __restrict__ arow,
                                               const _Float16* __restrict__ brow,
                                               int g) {
    v8f acc = {};
    for (int k0 = 0; k0 < EE; k0 += 32) {
        v16h a, bfrag;
        a = *(const v16h*)(arow + k0 + g * 16);
        bfrag = *(const v16h*)(brow + k0 + g * 16);
        acc = __builtin_amdgcn_wmma_f32_16x16x32_f16(
            /*neg_a=*/false, a, /*neg_b=*/false, bfrag,
            /*c_mod=*/(short)0, acc, /*reuse_a=*/false, /*reuse_b=*/false);
    }
    return acc;
}

// GEMM1 (+bias+SELU) -> f16. 8 waves/block; all waves share one
// TDM-staged B panel (n-tile) and cover 8 consecutive m-tiles.
// grid = (M/16/8) * 32 n-tiles.
__global__ void k_gemm_selu_h(const _Float16* __restrict__ A, const _Float16* __restrict__ Bt,
                              const float* __restrict__ bias, _Float16* __restrict__ C) {
    __shared__ _Float16 ldsB[16 * EE];   // 16 KB shared B panel
    int t = threadIdx.x, lane = t & 31, wave = t >> 5;
    int n0 = (blockIdx.x & 31) * 16;                 // n-tile
    int m0 = ((blockIdx.x >> 5) * 8 + wave) * 16;    // m-tile per wave
    if (wave == 0) {
        tdm_load_panel16(Bt + (size_t)n0 * EE, ldsB);
        __builtin_amdgcn_s_wait_tensorcnt(0);
    }
    __syncthreads();
    int g = lane >> 4, r = lane & 15;
    v8f acc = wmma_tile_mixed(A + (size_t)(m0 + r) * EE, ldsB + (size_t)r * EE, g);
    float bn = bias[n0 + r];
#pragma unroll
    for (int i = 0; i < 8; ++i)
        C[(size_t)(m0 + g * 8 + i) * EE + n0 + r] = (_Float16)selu_f(acc[i] + bn);
}

// GEMM2 (+bias+SELU) -> f32. M=64 -> 4 waves/block (one m-tile each),
// one TDM-staged B panel per block; grid = 32 n-tiles.
__global__ void k_gemm_selu_f(const _Float16* __restrict__ A, const _Float16* __restrict__ Bt,
                              const float* __restrict__ bias, float* __restrict__ C) {
    __shared__ _Float16 ldsB[16 * EE];
    int t = threadIdx.x, lane = t & 31, wave = t >> 5;
    int n0 = blockIdx.x * 16;
    int m0 = wave * 16;
    if (wave == 0) {
        tdm_load_panel16(Bt + (size_t)n0 * EE, ldsB);
        __builtin_amdgcn_s_wait_tensorcnt(0);
    }
    __syncthreads();
    int g = lane >> 4, r = lane & 15;
    v8f acc = wmma_tile_mixed(A + (size_t)(m0 + r) * EE, ldsB + (size_t)r * EE, g);
    float bn = bias[n0 + r];
#pragma unroll
    for (int i = 0; i < 8; ++i)
        C[(size_t)(m0 + g * 8 + i) * EE + n0 + r] = selu_f(acc[i] + bn);
}

// ---------------------------------------------------------------
// Masked mean of sent_h over the 32 sentences of a review.
// ---------------------------------------------------------------
__global__ void k_rev_mean(const _Float16* __restrict__ sh, const float* __restrict__ smask,
                           _Float16* __restrict__ rmean) {
    int rv = blockIdx.x;
    int c  = threadIdx.x * 2;
    float a0 = 0.f, a1 = 0.f, ms = 0.f;
    for (int s = 0; s < SS; ++s) {
        float m = smask[rv * SS + s];
        ms += m;
        const _Float16* p = sh + (size_t)(rv * SS + s) * EE + c;
        a0 += m * (float)p[0];
        a1 += m * (float)p[1];
    }
    float inv = 1.f / fmaxf(ms, 1.f);
    rmean[(size_t)rv * EE + c]     = (_Float16)(a0 * inv);
    rmean[(size_t)rv * EE + c + 1] = (_Float16)(a1 * inv);
}

// ---------------------------------------------------------------
// Fused tail (<1 MFLOP): user-feat L2 norm/scale, per-review star
// head, product mean, product encoder (532x532), label head.
// out[0..8] = p_stars, out[9..72] = r_stars (tuple concat order).
// ---------------------------------------------------------------
__global__ void k_final(const float* __restrict__ rev_h,
                        const float* __restrict__ user_feats, const float* __restrict__ user_w,
                        const float* __restrict__ w_prod, const float* __restrict__ b_prod,
                        const float* __restrict__ w_rff,  const float* __restrict__ b_rff,
                        const float* __restrict__ w_pff,  const float* __restrict__ b_pff,
                        float* __restrict__ out) {
    __shared__ float uf[RR][UU];
    __shared__ float pmean[DD];
    __shared__ float hn[DD];
    int t = threadIdx.x;

    if (t < RR) {   // uf[r] = user_feats[r]/max(||.||,1e-12) * user_w
        float nrm = 0.f;
        for (int u = 0; u < UU; ++u) { float x = user_feats[t * UU + u]; nrm += x * x; }
        float inv = 1.f / fmaxf(sqrtf(nrm), 1e-12f);
        for (int u = 0; u < UU; ++u) uf[t][u] = user_feats[t * UU + u] * inv * user_w[u];
    }
    __syncthreads();

    if (t < RR) {   // r_stars
        float acc = b_rff[0];
        for (int d = 0; d < EE; ++d) acc += rev_h[(size_t)t * EE + d] * w_rff[d];
        for (int u = 0; u < UU; ++u) acc += uf[t][u] * w_rff[EE + u];
        out[LL + t] = selu_f(acc);
    }

    for (int d = t; d < DD; d += 256) {   // pmean (plain mean over reviews)
        float s = 0.f;
        if (d < EE) { for (int r = 0; r < RR; ++r) s += rev_h[(size_t)r * EE + d]; }
        else        { for (int r = 0; r < RR; ++r) s += uf[r][d - EE]; }
        pmean[d] = s * (1.f / (float)RR);
    }
    __syncthreads();

    for (int j = t; j < DD; j += 256) {   // h_n = selu(pmean @ w_prod + b_prod)
        float s = b_prod[j];
        for (int d = 0; d < DD; ++d) s += pmean[d] * w_prod[(size_t)d * DD + j];
        hn[j] = selu_f(s);
    }
    __syncthreads();

    if (t < LL) {   // p_stars
        float s = b_pff[t];
        for (int j = 0; j < DD; ++j) s += hn[j] * w_pff[(size_t)j * LL + t];
        out[t] = selu_f(s);
    }
}

// ---------------- host-side launch ----------------
extern "C" void kernel_launch(void* const* d_in, const int* in_sizes, int n_in,
                              void* d_out, int out_size, void* d_ws, size_t ws_size,
                              hipStream_t stream) {
    const int*   inputs     = (const int*)d_in[0];    // [R,S,T]
    const float* user_feats = (const float*)d_in[1];  // [R,U]
    const float* embedding  = (const float*)d_in[2];  // [V,E]
    const float* w_sent     = (const float*)d_in[3];  // [E,E]
    const float* b_sent     = (const float*)d_in[4];  // [E]
    const float* w_rev      = (const float*)d_in[5];  // [E,E]
    const float* b_rev      = (const float*)d_in[6];  // [E]
    const float* w_prod     = (const float*)d_in[7];  // [D,D]
    const float* b_prod     = (const float*)d_in[8];  // [D]
    const float* w_rff      = (const float*)d_in[9];  // [D,1]
    const float* b_rff      = (const float*)d_in[10]; // [1]
    const float* w_pff      = (const float*)d_in[11]; // [D,L]
    const float* b_pff      = (const float*)d_in[12]; // [L]
    const float* user_w     = (const float*)d_in[13]; // [U]
    float* out = (float*)d_out;                       // 9 + 64 = 73 floats

    // workspace layout (all offsets 256B-aligned)
    char* ws = (char*)d_ws;
    _Float16* wtS   = (_Float16*)(ws + 0);        // 512*512 f16 transposed w_sent
    _Float16* wtR   = (_Float16*)(ws + 524288);   // 512*512 f16 transposed w_rev
    _Float16* smean = (_Float16*)(ws + 1048576);  // [2048,512] f16 sentence token-means
    float*    smask = (float*)   (ws + 3145728);  // [2048] sentence masks
    _Float16* sh    = (_Float16*)(ws + 3153920);  // [2048,512] f16 sent_h
    _Float16* rmean = (_Float16*)(ws + 5251072);  // [64,512]  f16 review means
    float*    rvh   = (float*)   (ws + 5316608);  // [64,512]  f32 rev_h

    k_convert_w<<<256, 256, 0, stream>>>(w_sent, wtS);
    k_convert_w<<<256, 256, 0, stream>>>(w_rev,  wtR);
    k_tok_mean<<<RR * SS, 256, 0, stream>>>(inputs, embedding, smean, smask);
    // GEMM1: 128 m-tiles / 8 per block * 32 n-tiles = 512 blocks, 8 waves
    k_gemm_selu_h<<<512, 256, 0, stream>>>(smean, wtS, b_sent, sh);
    k_rev_mean<<<RR, 256, 0, stream>>>(sh, smask, rmean);
    // GEMM2: 4 m-tiles (4 waves) * 32 n-tile blocks
    k_gemm_selu_f<<<32, 128, 0, stream>>>(rmean, wtR, b_rev, rvh);
    k_final<<<1, 256, 0, stream>>>(rvh, user_feats, user_w,
                                   w_prod, b_prod, w_rff, b_rff, w_pff, b_pff, out);
}
